// EmbeddingBridge_46531675685395
// MI455X (gfx1250) — matrix operand order repaired and checked by reading, not compiled, not run on hardware
//
#include <hip/hip_runtime.h>
#include <hip/hip_bf16.h>

// ---------------------------------------------------------------------------
// Cosine-similarity argmax over vocab, MI455X (gfx1250, wave32, WMMA).
//   latent  [512, 768] f32 , embedding [250002, 768] f32 -> token ids [512] i32
// One HBM pass over the 768MB embedding; bf16 v_wmma_f32_16x16x32_bf16 GEMM;
// fused norms + argmax via packed u64 atomic max.
// Per-wave tile 32x64 (2 Msub x 4 Nsub = 64 accum VGPRs); M=512 covered in two
// passes over the LDS B tile. A __syncthreads() fence per pass prevents LICM
// from hoisting the LDS B-fragment loads across passes (which caused spills).
// ---------------------------------------------------------------------------

typedef __attribute__((ext_vector_type(16))) __bf16 v16bf;
typedef __attribute__((ext_vector_type(8)))  float  v8f;

#define KDIM   768
#define KSTEPS 24           // 768 / 32
#define NTILE  64           // vocab rows per workgroup
#define BFRAG_BYTES (4 * KSTEPS * 1024)   // 96 KB of B fragments in LDS

static __device__ __forceinline__ unsigned short f2bf(float f) {
  unsigned u = __float_as_uint(f);
  unsigned r = u + 0x7FFFu + ((u >> 16) & 1u);   // round-to-nearest-even
  return (unsigned short)(r >> 16);
}

static __device__ __forceinline__ unsigned f2ord(float f) {
  unsigned u = __float_as_uint(f);
  return (u & 0x80000000u) ? ~u : (u | 0x80000000u);  // order-preserving map
}

static __device__ __forceinline__ unsigned long long shflxor_u64(unsigned long long v, int m) {
  int lo = __shfl_xor((int)(unsigned)v, m, 32);
  int hi = __shfl_xor((int)(unsigned)(v >> 32), m, 32);
  return ((unsigned long long)(unsigned)hi << 32) | (unsigned)lo;
}

// ---------------------------------------------------------------------------
// Kernel 1: normalize latent rows (fp32), emit bf16 in WMMA A-fragment layout.
// A 16x32 bf16 fragment (ISA 7.12.2): lanes 0-15 row M=l hold K 0..7 (V0..V3)
// and 16..23 (V4..V7); lanes 16-31 row M=l-16 hold K 8..15 / 24..31.
// Fragment (msub, kstep) stored as 32 lanes x 32 bytes = 1KB contiguous.
// Also zeroes the per-row best[] array (graph-replay safe re-init).
// ---------------------------------------------------------------------------
__global__ void __launch_bounds__(256)
pack_latent_kernel(const float* __restrict__ lat,
                   unsigned long long* __restrict__ best,
                   unsigned short* __restrict__ wsA, int M) {
  const int tid  = threadIdx.x;
  const int lane = tid & 31;
  const int w    = tid >> 5;

  if (blockIdx.x == 0) {                  // re-init merge array every launch
    for (int j = tid; j < M; j += blockDim.x) best[j] = 0ull;
  }

  const int m = blockIdx.x * 8 + w;       // one wave per latent row
  if (m >= M) return;
  const float* row = lat + (size_t)m * KDIM;

  float x[24];
  float ss = 0.f;
#pragma unroll
  for (int i = 0; i < 12; ++i) {          // pair index kp = lane + 32*i
    float2 v = *(const float2*)(row + 2 * (lane + 32 * i));
    x[2 * i] = v.x; x[2 * i + 1] = v.y;
    ss += v.x * v.x + v.y * v.y;
  }
#pragma unroll
  for (int off = 16; off >= 1; off >>= 1) ss += __shfl_xor(ss, off, 32);
  const float inv = 1.f / fmaxf(sqrtf(ss), 1e-12f);

  const int msub = m >> 4, mrow = m & 15;
#pragma unroll
  for (int i = 0; i < 12; ++i) {
    const int kp = lane + 32 * i;
    const int k  = kp * 2;                 // even -> (k,k+1) share one dword
    const int kstep = k >> 5, kk = k & 31;
    const int lane_half = (kk >> 3) & 1;
    const int vg = ((kk >> 4) << 2) | ((kk & 7) >> 1);
    const unsigned pk = (unsigned)f2bf(x[2 * i] * inv) |
                        ((unsigned)f2bf(x[2 * i + 1] * inv) << 16);
    const size_t off = ((size_t)(msub * KSTEPS + kstep) << 10) +
                       (size_t)(mrow + 16 * lane_half) * 32 + (size_t)vg * 4;
    *(unsigned*)((char*)wsA + off) = pk;
  }
}

// ---------------------------------------------------------------------------
// Kernel 2: per 64-row vocab tile: stream fp32 -> bf16 B-fragments in LDS
// (+ on-the-fly row sum-of-squares), then 8 waves x 2 passes of
// (2 Msub x 4 Nsub) WMMA over K=768, scale by 1/||e||, reduce to packed
// (value,~idx) u64 keys, global_atomic_max_u64 into best[].
// B 32x16 bf16 fragment: lanes 0-15 col N=l hold K 0..7 (V0..V3) and 16..23 (V4..V7),
// lanes 16-31 col N=l-16 hold K 8..15 / 24..31.
// ---------------------------------------------------------------------------
__global__ void __launch_bounds__(256)
gemm_argmax_kernel(const float* __restrict__ emb,
                   const unsigned short* __restrict__ wsA,
                   unsigned long long* __restrict__ best,
                   int V) {
  extern __shared__ char smem[];                 // [0,96K): B frags; then 64 f32
  float* sumsq = (float*)(smem + BFRAG_BYTES);

  const int tid  = threadIdx.x;
  const int lane = tid & 31;
  const int w    = tid >> 5;                     // 8 waves
  const long nbase = (long)blockIdx.x * NTILE;

  if (tid < NTILE) sumsq[tid] = 0.f;
  __syncthreads();

  // ---- stage B tile: wave w handles vocab rows [w*8, w*8+8) of the tile ----
#pragma unroll
  for (int rr = 0; rr < 8; ++rr) {
    const int  r    = w * 8 + rr;
    const long vrow = nbase + r;
    const bool ok   = vrow < V;
    const float* src = emb + (size_t)vrow * KDIM;
    const int nsub = r >> 4, n = r & 15;
    float ss = 0.f;
#pragma unroll
    for (int i = 0; i < 12; ++i) {
      const int k = 2 * (i * 32 + lane);         // coalesced float2 per lane
      float2 v = ok ? *(const float2*)(src + k) : float2{0.f, 0.f};
      ss += v.x * v.x + v.y * v.y;
      const int kstep = k >> 5, kk = k & 31;
      const int bl = n + 16 * ((kk >> 3) & 1);
      const int vg = ((kk >> 4) << 2) | ((kk & 7) >> 1);
      const unsigned pk = (unsigned)f2bf(v.x) | ((unsigned)f2bf(v.y) << 16);
      *(unsigned*)(smem + (((nsub * KSTEPS + kstep) << 10) + bl * 32 + vg * 4)) = pk;
    }
    atomicAdd(&sumsq[r], ss);
  }
  __syncthreads();
  if (tid < NTILE)                                // in-place: sumsq -> invnorm
    sumsq[tid] = 1.f / fmaxf(sqrtf(sumsq[tid]), 1e-12f);
  __syncthreads();

  // ---- per-column scale / index data (shared by both passes) ----
  const int nc = lane & 15;
  float invn[4]; unsigned nidx_inv[4]; bool okc[4];
#pragma unroll
  for (int nsub = 0; nsub < 4; ++nsub) {
    long idx = nbase + nsub * 16 + nc;
    invn[nsub] = sumsq[nsub * 16 + nc];
    okc[nsub]  = idx < V;
    nidx_inv[nsub] = ~(unsigned)idx;               // ties -> smallest index
  }

  const char* bBase = smem + lane * 32;

  // ---- two passes over M: pass h covers rows h*256 + w*32 .. +31 ----
#pragma unroll 1
  for (int h = 0; h < 2; ++h) {
    // Memory fence: forces this pass to issue its own ds_load_b128s instead of
    // LICM keeping 96 B-fragments (768 VGPRs) live across passes -> spills.
    __syncthreads();

    v8f acc[2][4];
#pragma unroll
    for (int i = 0; i < 2; ++i)
#pragma unroll
      for (int j = 0; j < 4; ++j)
        acc[i][j] = v8f{0.f,0.f,0.f,0.f,0.f,0.f,0.f,0.f};

    const int msub_g0 = h * 16 + w * 2;            // global 16-row M subtile
    const char* aBase = (const char*)wsA +
        ((size_t)msub_g0 * KSTEPS << 10) + (size_t)lane * 32;

#pragma unroll 4
    for (int ks = 0; ks < KSTEPS; ++ks) {
#define WMMA(M_, A_, N_, B_) \
      acc[M_][N_] = __builtin_amdgcn_wmma_f32_16x16x32_bf16(false, A_, false, B_, \
                        (short)0, acc[M_][N_], false, false)
      v16bf a0 = *(const v16bf*)(aBase + ((0 * KSTEPS + ks) << 10));
      v16bf a1 = *(const v16bf*)(aBase + ((1 * KSTEPS + ks) << 10));
      {
        v16bf b0 = *(const v16bf*)(bBase + ((0 * KSTEPS + ks) << 10));
        v16bf b1 = *(const v16bf*)(bBase + ((1 * KSTEPS + ks) << 10));
        WMMA(0, a0, 0, b0); WMMA(1, a1, 0, b0);
        WMMA(0, a0, 1, b1); WMMA(1, a1, 1, b1);
      }
      {
        v16bf b2 = *(const v16bf*)(bBase + ((2 * KSTEPS + ks) << 10));
        v16bf b3 = *(const v16bf*)(bBase + ((3 * KSTEPS + ks) << 10));
        WMMA(0, a0, 2, b2); WMMA(1, a1, 2, b2);
        WMMA(0, a0, 3, b3); WMMA(1, a1, 3, b3);
      }
#undef WMMA
    }

    // ---- fused scale + argmax for this pass's 32 rows ----
#pragma unroll
    for (int msub = 0; msub < 2; ++msub) {
#pragma unroll
      for (int v = 0; v < 8; ++v) {
        unsigned long long kbest = 0ull;
#pragma unroll
        for (int nsub = 0; nsub < 4; ++nsub) {
          float val = acc[msub][nsub][v] * invn[nsub];
          unsigned long long key = okc[nsub]
              ? (((unsigned long long)f2ord(val) << 32) | nidx_inv[nsub]) : 0ull;
          kbest = key > kbest ? key : kbest;
        }
#pragma unroll
        for (int d = 1; d <= 8; d <<= 1) {         // reduce within 16-lane half
          unsigned long long o = shflxor_u64(kbest, d);
          kbest = o > kbest ? o : kbest;
        }
        if (nc == 0) {
          const int m = (msub_g0 + msub) * 16 + v + 8 * (lane >> 4);
          atomicMax(&best[m], kbest);              // global_atomic_max_u64
        }
      }
    }
  }
}

// ---------------------------------------------------------------------------
// Kernel 3: unpack token ids.
// ---------------------------------------------------------------------------
__global__ void finalize_kernel(const unsigned long long* __restrict__ best,
                                int* __restrict__ out, int M) {
  const int i = blockIdx.x * blockDim.x + threadIdx.x;
  if (i < M) out[i] = (int)(~(unsigned)(best[i] & 0xFFFFFFFFull));
}

extern "C" void kernel_launch(void* const* d_in, const int* in_sizes, int n_in,
                              void* d_out, int out_size, void* d_ws, size_t ws_size,
                              hipStream_t stream) {
  const float* latent = (const float*)d_in[0];
  const float* emb    = (const float*)d_in[1];
  int* out            = (int*)d_out;

  const int M = in_sizes[0] / KDIM;   // 512
  const int V = in_sizes[1] / KDIM;   // 250002

  unsigned long long* best = (unsigned long long*)d_ws;                  // M * 8B
  unsigned short* wsA = (unsigned short*)((char*)d_ws + 4096);           // M*768*2B

  pack_latent_kernel<<<(M + 7) / 8, 256, 0, stream>>>(latent, best, wsA, M);

  const int ntiles = (V + NTILE - 1) / NTILE;
  const size_t shmem = BFRAG_BYTES + NTILE * sizeof(float);
  gemm_argmax_kernel<<<ntiles, 256, shmem, stream>>>(emb, wsA, best, V);

  finalize_kernel<<<(M + 255) / 256, 256, 0, stream>>>(best, out, M);
}
